// DecoderBlock_58093727646336
// MI455X (gfx1250) — matrix operand order repaired and checked by reading, not compiled, not run on hardware
//
#include <hip/hip_runtime.h>

// ---------------------------------------------------------------------------
// Types / WMMA helpers (CDNA5 gfx1250, wave32)
// ---------------------------------------------------------------------------
typedef __attribute__((ext_vector_type(16))) __bf16 v16bf;
typedef __attribute__((ext_vector_type(4)))  __bf16 v4bf;
typedef __attribute__((ext_vector_type(8)))  float  v8f;
typedef __attribute__((ext_vector_type(4)))  float  v4f;

__device__ inline __bf16 f2bf(float f) {
  union { float f; unsigned u; } v; v.f = f;
  unsigned r = v.u + 0x7FFFu + ((v.u >> 16) & 1u);   // round-to-nearest-even
  unsigned short h = (unsigned short)(r >> 16);
  return __builtin_bit_cast(__bf16, h);
}

// Native packed f32->bf16 (RNE). gfx1250 is gfx12-family; v_cvt_pk_bf16_f32
// exists there even though the clang builtin is gated to gfx950.
__device__ inline uint2 f2bf4_raw(v4f f) {
  unsigned lo, hi;
  asm("v_cvt_pk_bf16_f32 %0, %1, %2" : "=v"(lo) : "v"(f[0]), "v"(f[1]));
  asm("v_cvt_pk_bf16_f32 %0, %1, %2" : "=v"(hi) : "v"(f[2]), "v"(f[3]));
  uint2 u; u.x = lo; u.y = hi;
  return u;
}

__device__ inline v4bf f2bf4(v4f f) {
  return __builtin_bit_cast(v4bf, f2bf4_raw(f));
}

__device__ inline v8f wmma_bf16(v16bf a, v16bf b, v8f c) {
  return __builtin_amdgcn_wmma_f32_16x16x32_bf16(false, a, false, b,
                                                 (short)0, c, false, false);
}

// A-fragment swizzle (16x32 tile): element K=c of row m lives at [m][kh][pos]
//   kh = c>>4 ; pos = c&15
// B-fragment swizzle (32x16 tile): element (k,n) lives at [kh][n][k&15]
// -> each lane's 16-half fragment is one contiguous 32-byte LDS read.

// A-frag from an f32 16x32 tile (flash probabilities), vectorized loads
__device__ inline v16bf load_a_f32v(const float* t, int lane) {
  int m  = lane & 15;
  int kh = (lane >> 4) & 1;
  const float* row = t + m * 32;
  uint2 q0 = f2bf4_raw(*(const v4f*)(row + kh * 16));
  uint2 q1 = f2bf4_raw(*(const v4f*)(row + kh * 16 + 4));
  uint2 q2 = f2bf4_raw(*(const v4f*)(row + kh * 16 + 8));
  uint2 q3 = f2bf4_raw(*(const v4f*)(row + kh * 16 + 12));
  struct P { uint2 a, b, c, d; } p{q0, q1, q2, q3};
  return __builtin_bit_cast(v16bf, p);
}

__device__ inline void lds_wait() {
  asm volatile("s_wait_dscnt 0" ::: "memory");
}

// ---------------------------------------------------------------------------
// Generic batched GEMM:  C[z] = act( alpha * (A[z] @ op(W[z])^T + bias) )
// Block tile 64x256, 8 waves (2x4), each wave 32x64 via 2x4 WMMA frags, BK=32.
// Per K-chunk per wave: 6 fragment loads -> 8 WMMAs.
// Requires: M multiple of 64, K multiple of 32 (true for all call sites).
// ---------------------------------------------------------------------------
#define BM 64
#define BN 256
#define BK 32

__global__ __launch_bounds__(256) void gemm_wmma(
    const float* __restrict__ A, const float* __restrict__ W,
    const float* __restrict__ bias, float* __restrict__ C,
    int M, int Nout, int K, int lda, int ldw, int ldc,
    long sAn, long sAh, long sWn, long sWh, long sCn, long sCh, long sBh,
    int Hb, float alpha, int relu, int transW)
{
  __shared__ __align__(32) __bf16 As2[4][16][2][16];    // [mt][m][kh][pos]  4KB
  __shared__ __align__(32) __bf16 Bs2[16][2][16][16];   // [cg][kh][n][kpos] 16KB

  int z = blockIdx.z;
  int bn = z / Hb, bh = z % Hb;
  A += bn * sAn + bh * sAh;
  W += bn * sWn + bh * sWh;
  C += bn * sCn + bh * sCh;
  if (bias) bias += bh * sBh;

  int m0 = blockIdx.y * BM;
  int n0 = blockIdx.x * BN;
  int tid  = threadIdx.x;
  int lane = tid & 31, wave = tid >> 5;
  int wm = (wave >> 2) * 32;        // 0 / 32
  int wn = (wave & 3) * 64;         // 0 / 64 / 128 / 192

  int fl_m  = lane & 15;
  int fl_kh = (lane >> 4) & 1;
  bool bfast = (n0 + BN <= Nout) && !transW;

  // hoisted staging addresses (A: 2 vec4/thread, B fast: 8 vec4/thread)
  int ac4 = (tid & 7) * 4;
  int ar0 = tid >> 3;                                     // 0..31
  const float* Arow0 = A + (long)(m0 + ar0) * lda + ac4;
  const float* Arow1 = Arow0 + (long)32 * lda;
  int akh  = ac4 >> 4;
  int apos = ac4 & 15;
  __bf16* as0 = &As2[(ar0 >> 4) + 0][ar0 & 15][akh][apos];
  __bf16* as1 = &As2[(ar0 >> 4) + 2][ar0 & 15][akh][apos];

  int bk4 = (tid & 7) * 4;
  int bn0t = tid >> 3;                                    // 0..31
  const float* Wrow = W + (long)(n0 + bn0t) * ldw + bk4;

  v8f acc[2][4] = {};

  for (int kc = 0; kc < K; kc += BK) {
    // ---- stage A (always full tile) ----
    *(v4bf*)as0 = f2bf4(*(const v4f*)(Arow0 + kc));
    *(v4bf*)as1 = f2bf4(*(const v4f*)(Arow1 + kc));
    if (kc + BK < K) {
      __builtin_prefetch(Arow0 + kc + BK, 0, 3);
      __builtin_prefetch(Arow1 + kc + BK, 0, 3);
    }
    // ---- stage B ----
    if (bfast) {
#pragma unroll
      for (int it = 0; it < 8; ++it) {
        int n = bn0t + it * 32;
        v4f f = *(const v4f*)(Wrow + (long)it * 32 * ldw + kc);
        *(v4bf*)(&Bs2[n >> 4][bk4 >> 4][n & 15][bk4 & 15]) = f2bf4(f);
      }
    } else {
      for (int i = tid; i < BK * BN; i += 256) {
        int r = i >> 8, c = i & 255;                     // r=k, c=n
        int gk = kc + r, gn = n0 + c;
        float v = 0.f;
        if (gk < K && gn < Nout)
          v = transW ? W[(long)gk * ldw + gn] : W[(long)gn * ldw + gk];
        Bs2[c >> 4][r >> 4][c & 15][r & 15] = f2bf(v);
      }
    }
    __syncthreads();

    v16bf a0 = *(const v16bf*)(&As2[(wm >> 4) + 0][fl_m][fl_kh][0]);
    v16bf a1 = *(const v16bf*)(&As2[(wm >> 4) + 1][fl_m][fl_kh][0]);
#pragma unroll
    for (int j = 0; j < 4; ++j) {
      v16bf b = *(const v16bf*)(&Bs2[(wn >> 4) + j][fl_kh][fl_m][0]);
      acc[0][j] = wmma_bf16(a0, b, acc[0][j]);
      acc[1][j] = wmma_bf16(a1, b, acc[1][j]);
    }
    __syncthreads();
  }

  int cn = lane & 15;
  int cm = ((lane >> 4) & 1) * 8;
#pragma unroll
  for (int i = 0; i < 2; ++i)
#pragma unroll
    for (int j = 0; j < 4; ++j) {
      int col = n0 + wn + j * 16 + cn;
      if (col < Nout) {
        float bsv = bias ? bias[col] : 0.f;
#pragma unroll
        for (int r = 0; r < 8; ++r) {
          int row = m0 + wm + i * 16 + cm + r;
          float v = (acc[i][j][r] + bsv) * alpha;
          if (relu) v = fmaxf(v, 0.f);
          C[(long)row * ldc + col] = v;
        }
      }
    }
}

// ---------------------------------------------------------------------------
// Causal flash attention (self-attn). One wave per block, 16 query rows.
// ---------------------------------------------------------------------------
__global__ __launch_bounds__(32) void flash_attn_causal(
    const float* __restrict__ Q, const float* __restrict__ K,
    const float* __restrict__ V, float* __restrict__ O,
    int Lseq, int Estride, float scale)
{
  __shared__ __align__(32) __bf16 QS2[2][16][2][16];     // A-frags per k-chunk
  __shared__ __align__(32) __bf16 KB[2][2][2][16][16];   // [dchunk][keygrp][kh][n][kpos]
  __shared__ __align__(32) __bf16 VB[4][2][16][16];      // [dcg][kh][n][kpos]
  __shared__ __align__(32) float  PS[16][32];

  int lane  = threadIdx.x;
  int qtile = blockIdx.x;
  int h     = blockIdx.y;
  int n     = blockIdx.z;
  long base = ((long)n * Lseq) * Estride + h * 64;

  int fl_m  = lane & 15;
  int fl_kh = (lane >> 4) & 1;

  // ---- stage Q (16x64) as two A-fragment tiles ----
  {
    int c4 = (lane & 15) * 4;
    int r0 = lane >> 4;                            // 0..1
    int cc  = c4 & 31;
    int kh  = cc >> 4;
    int pos = cc & 15;
    const float* qp = Q + base + (long)(qtile * 16 + r0) * Estride + c4;
#pragma unroll
    for (int it = 0; it < 8; ++it) {
      int r = r0 + it * 2;
      v4f f = *(const v4f*)(qp + (long)it * 2 * Estride);
      *(v4bf*)(&QS2[c4 >> 5][r][kh][pos]) = f2bf4(f);
    }
  }
  lds_wait();
  v16bf aq0 = *(const v16bf*)(&QS2[0][fl_m][fl_kh][0]);
  v16bf aq1 = *(const v16bf*)(&QS2[1][fl_m][fl_kh][0]);

  v8f o0 = {}, o1 = {}, o2 = {}, o3 = {};
  float mrow[8], lrow[8];
#pragma unroll
  for (int r = 0; r < 8; ++r) { mrow[r] = -3.0e38f; lrow[r] = 0.f; }

  int cn = lane & 15;
  int cm = ((lane >> 4) & 1) * 8;
  int ntiles = ((qtile * 16 + 15) >> 5) + 1;

  int c4  = (lane & 15) * 4;                       // d offset this lane stages
  int kk0 = lane >> 4;                             // 0..1
  const float* kp = K + base + c4;
  const float* vp = V + base + c4;
  int kcol = c4 & 31;

  for (int t = 0; t < ntiles; ++t) {
    lds_wait();   // previous iteration's frag reads done before refill
    {
      long rowb = (long)(t * 32 + kk0) * Estride;
#pragma unroll
      for (int it = 0; it < 16; ++it) {
        int kk = kk0 + it * 2;
        long off = rowb + (long)it * 2 * Estride;
        v4f fk = *(const v4f*)(kp + off);
        v4f fv = *(const v4f*)(vp + off);
        *(v4bf*)(&KB[c4 >> 5][kk >> 4][kcol >> 4][kk & 15][kcol & 15]) = f2bf4(fk);
        v4bf hv = f2bf4(fv);
#pragma unroll
        for (int j = 0; j < 4; ++j) {
          int d = c4 + j;
          VB[d >> 4][kk >> 4][d & 15][kk & 15] = hv[j];
        }
      }
    }
    lds_wait();

    v8f s0 = {}, s1 = {};
    {
      v16bf b00 = *(const v16bf*)(&KB[0][0][fl_kh][fl_m][0]);
      v16bf b10 = *(const v16bf*)(&KB[1][0][fl_kh][fl_m][0]);
      s0 = wmma_bf16(aq0, b00, s0);
      s0 = wmma_bf16(aq1, b10, s0);
      v16bf b01 = *(const v16bf*)(&KB[0][1][fl_kh][fl_m][0]);
      v16bf b11 = *(const v16bf*)(&KB[1][1][fl_kh][fl_m][0]);
      s1 = wmma_bf16(aq0, b01, s1);
      s1 = wmma_bf16(aq1, b11, s1);
    }

    int k0 = t * 32 + cn, k1 = t * 32 + 16 + cn;
#pragma unroll
    for (int r = 0; r < 8; ++r) {
      int qg = qtile * 16 + cm + r;
      float v0 = (k0 <= qg) ? s0[r] * scale : -3.0e18f;
      float v1 = (k1 <= qg) ? s1[r] * scale : -3.0e18f;
      float rmax = fmaxf(v0, v1);
#pragma unroll
      for (int off = 8; off > 0; off >>= 1)
        rmax = fmaxf(rmax, __shfl_xor(rmax, off, 32));
      float nm = fmaxf(mrow[r], rmax);
      float p0 = __expf(v0 - nm);
      float p1 = __expf(v1 - nm);
      float rsum = p0 + p1;
#pragma unroll
      for (int off = 8; off > 0; off >>= 1)
        rsum += __shfl_xor(rsum, off, 32);
      float corr = __expf(mrow[r] - nm);
      lrow[r] = lrow[r] * corr + rsum;
      mrow[r] = nm;
      o0[r] *= corr; o1[r] *= corr; o2[r] *= corr; o3[r] *= corr;
      PS[cm + r][cn]      = p0;
      PS[cm + r][16 + cn] = p1;
    }
    lds_wait();

    v16bf ap  = load_a_f32v(&PS[0][0], lane);
    v16bf bv0 = *(const v16bf*)(&VB[0][fl_kh][fl_m][0]);
    v16bf bv1 = *(const v16bf*)(&VB[1][fl_kh][fl_m][0]);
    v16bf bv2 = *(const v16bf*)(&VB[2][fl_kh][fl_m][0]);
    v16bf bv3 = *(const v16bf*)(&VB[3][fl_kh][fl_m][0]);
    o0 = wmma_bf16(ap, bv0, o0);
    o1 = wmma_bf16(ap, bv1, o1);
    o2 = wmma_bf16(ap, bv2, o2);
    o3 = wmma_bf16(ap, bv3, o3);
  }

#pragma unroll
  for (int r = 0; r < 8; ++r) {
    float inv = 1.0f / lrow[r];
    long row = base + (long)(qtile * 16 + cm + r) * Estride;
    O[row + 0  + cn] = o0[r] * inv;
    O[row + 16 + cn] = o1[r] * inv;
    O[row + 32 + cn] = o2[r] * inv;
    O[row + 48 + cn] = o3[r] * inv;
  }
}

// ---------------------------------------------------------------------------
// Block-wide reductions (block = 256 threads = 8 waves)
// ---------------------------------------------------------------------------
__device__ inline float block_sum(float v, float* sh) {
#pragma unroll
  for (int o = 16; o > 0; o >>= 1) v += __shfl_xor(v, o, 32);
  __syncthreads();
  if ((threadIdx.x & 31) == 0) sh[threadIdx.x >> 5] = v;
  __syncthreads();
  float r = 0.f;
#pragma unroll
  for (int i = 0; i < 8; ++i) r += sh[i];
  return r;
}

__device__ inline float block_max(float v, float* sh) {
#pragma unroll
  for (int o = 16; o > 0; o >>= 1) v = fmaxf(v, __shfl_xor(v, o, 32));
  __syncthreads();
  if ((threadIdx.x & 31) == 0) sh[threadIdx.x >> 5] = v;
  __syncthreads();
  float r = sh[0];
#pragma unroll
  for (int i = 1; i < 8; ++i) r = fmaxf(r, sh[i]);
  return r;
}

// Row softmax, cols == 1024: whole row held in registers; 1 read + 1 write.
__global__ __launch_bounds__(256) void softmax_row1024(float* __restrict__ a) {
  __shared__ float sh[8];
  float* p = a + (long)blockIdx.x * 1024;
  int c4 = threadIdx.x * 4;
  v4f v = *(const v4f*)(p + c4);
  float mx = fmaxf(fmaxf(v[0], v[1]), fmaxf(v[2], v[3]));
  mx = block_max(mx, sh);
  v4f e;
  float s = 0.f;
#pragma unroll
  for (int j = 0; j < 4; ++j) { e[j] = __expf(v[j] - mx); s += e[j]; }
  s = block_sum(s, sh);
  float inv = 1.f / s;
#pragma unroll
  for (int j = 0; j < 4; ++j) e[j] *= inv;
  *(v4f*)(p + c4) = e;
}

// y = LayerNorm(a + b) * g + beta, E == 1024, row in registers.
__global__ __launch_bounds__(256) void add_ln1024(
    const float* __restrict__ a, const float* __restrict__ b,
    const float* __restrict__ g, const float* __restrict__ beta,
    float* __restrict__ y)
{
  __shared__ float sh[8];
  long row = (long)blockIdx.x * 1024;
  int c4 = threadIdx.x * 4;
  v4f va = *(const v4f*)(a + row + c4);
  v4f vb = *(const v4f*)(b + row + c4);
  v4f t;
  float s = 0.f, s2 = 0.f;
#pragma unroll
  for (int j = 0; j < 4; ++j) { t[j] = va[j] + vb[j]; s += t[j]; s2 += t[j] * t[j]; }
  s  = block_sum(s,  sh);
  s2 = block_sum(s2, sh);
  float mean = s * (1.f / 1024.f);
  float var  = s2 * (1.f / 1024.f) - mean * mean;
  float rstd = rsqrtf(var + 1e-5f);
  v4f vg  = *(const v4f*)(g + c4);
  v4f vbe = *(const v4f*)(beta + c4);
  v4f o;
#pragma unroll
  for (int j = 0; j < 4; ++j) o[j] = (t[j] - mean) * rstd * vg[j] + vbe[j];
  *(v4f*)(y + row + c4) = o;
}

// ---------------------------------------------------------------------------
// Host: full decoder block pipeline
// ---------------------------------------------------------------------------
extern "C" void kernel_launch(void* const* d_in, const int* in_sizes, int n_in,
                              void* d_out, int out_size, void* d_ws, size_t ws_size,
                              hipStream_t stream) {
  const int Nn = 4, Ls = 1024, Ee = 1024, Hh = 16, Ff = 4096;
  const int Mr = Nn * Ls;
  const long SZ = (long)Mr * Ee;

  const float* x       = (const float*)d_in[0];
  const float* value   = (const float*)d_in[1];
  const float* key_inp = (const float*)d_in[2];
  const float* Wq  = (const float*)d_in[3];
  const float* bq  = (const float*)d_in[4];
  const float* Wk  = (const float*)d_in[5];
  const float* bk  = (const float*)d_in[6];
  const float* Wv  = (const float*)d_in[7];
  const float* bv  = (const float*)d_in[8];
  const float* Wo  = (const float*)d_in[9];
  const float* bo  = (const float*)d_in[10];
  const float* ln_g = (const float*)d_in[11];
  const float* ln_b = (const float*)d_in[12];
  const float* in_w = (const float*)d_in[13];
  const float* in_b = (const float*)d_in[14];
  const float* ow  = (const float*)d_in[15];
  const float* ob  = (const float*)d_in[16];
  const float* ln1_g = (const float*)d_in[17];
  const float* ln1_b = (const float*)d_in[18];
  const float* ln2_g = (const float*)d_in[19];
  const float* ln2_b = (const float*)d_in[20];
  const float* W1  = (const float*)d_in[21];
  const float* b1  = (const float*)d_in[22];
  const float* W2  = (const float*)d_in[23];
  const float* b2  = (const float*)d_in[24];

  float* out   = (float*)d_out;              // (N, L, E)
  float* attnW = out + SZ;                   // (N, H, L, L)

  float* ws  = (float*)d_ws;
  float* qb  = ws;            // q  / qm
  float* kb  = ws + 1 * SZ;   // k  / km
  float* vb  = ws + 2 * SZ;   // v  / vm
  float* sa  = ws + 3 * SZ;   // self-attn out / cross ctx
  float* da  = ws + 4 * SZ;   // dec_attention / mha
  float* dq  = ws + 5 * SZ;   // dec_query
  float* x2b = ws + 6 * SZ;   // x2
  float* ff1 = ws + 7 * SZ;   // (N, L, F)  -> 4*SZ
  float* ff2 = ws + 11 * SZ;  // (N, L, E)

  // 1-3. per-head q/k/v projections
  gemm_wmma<<<dim3(1, 64, 16), 256, 0, stream>>>(
      x, Wq, bq, qb, Mr, 64, 64, Ee, 64, Ee,
      0, 64, 0, 64 * 64, 0, 64, 64, Hh, 1.0f, 0, 0);
  gemm_wmma<<<dim3(1, 64, 16), 256, 0, stream>>>(
      x, Wk, bk, kb, Mr, 64, 64, Ee, 64, Ee,
      0, 64, 0, 64 * 64, 0, 64, 64, Hh, 1.0f, 0, 0);
  gemm_wmma<<<dim3(1, 64, 16), 256, 0, stream>>>(
      x, Wv, bv, vb, Mr, 64, 64, Ee, 64, Ee,
      0, 64, 0, 64 * 64, 0, 64, 64, Hh, 1.0f, 0, 0);

  // 4. causal flash self-attention (scale = 1/sqrt(E))
  flash_attn_causal<<<dim3(Ls / 16, Hh, Nn), 32, 0, stream>>>(
      qb, kb, vb, sa, Ls, Ee, 0.03125f);

  // 5. dec_attention = sa @ Wo^T + bo
  gemm_wmma<<<dim3(4, 64, 1), 256, 0, stream>>>(
      sa, Wo, bo, da, Mr, Ee, Ee, Ee, Ee, Ee,
      0, 0, 0, 0, 0, 0, 0, 1, 1.0f, 0, 0);

  // 6. dec_query = LN(dec_attention + x)
  add_ln1024<<<Mr, 256, 0, stream>>>(da, x, ln_g, ln_b, dq);

  // 7-9. cross projections (qm scaled by D^-0.5 after bias)
  gemm_wmma<<<dim3(4, 64, 1), 256, 0, stream>>>(
      value, in_w, in_b, qb, Mr, Ee, Ee, Ee, Ee, Ee,
      0, 0, 0, 0, 0, 0, 0, 1, 0.125f, 0, 0);
  gemm_wmma<<<dim3(4, 64, 1), 256, 0, stream>>>(
      key_inp, in_w + (long)Ee * Ee, in_b + Ee, kb, Mr, Ee, Ee, Ee, Ee, Ee,
      0, 0, 0, 0, 0, 0, 0, 1, 1.0f, 0, 0);
  gemm_wmma<<<dim3(4, 64, 1), 256, 0, stream>>>(
      dq, in_w + 2L * Ee * Ee, in_b + 2 * Ee, vb, Mr, Ee, Ee, Ee, Ee, Ee,
      0, 0, 0, 0, 0, 0, 0, 1, 1.0f, 0, 0);

  // 10. scores[n,h] = qm[n,:,h,:] @ km[n,:,h,:]^T  -> attnW (in d_out)
  gemm_wmma<<<dim3(4, 16, Nn * Hh), 256, 0, stream>>>(
      qb, kb, nullptr, attnW, Ls, Ls, 64, Ee, Ee, Ls,
      (long)Ls * Ee, 64, (long)Ls * Ee, 64,
      (long)Hh * Ls * Ls, (long)Ls * Ls, 0, Hh, 1.0f, 0, 0);

  // 11. attn_w = softmax(scores), in place (1 read + 1 write pass)
  softmax_row1024<<<Nn * Hh * Ls, 256, 0, stream>>>(attnW);

  // 12. ctx[n,:,h,:] = attn_w[n,h] @ vm[n,:,h,:]
  gemm_wmma<<<dim3(1, 16, Nn * Hh), 256, 0, stream>>>(
      attnW, vb, nullptr, sa, Ls, 64, Ls, Ls, Ee, Ee,
      (long)Hh * Ls * Ls, (long)Ls * Ls, (long)Ls * Ee, 64,
      (long)Ls * Ee, 64, 0, Hh, 1.0f, 0, 1);

  // 13. mha = ctx @ ow^T + ob
  gemm_wmma<<<dim3(4, 64, 1), 256, 0, stream>>>(
      sa, ow, ob, da, Mr, Ee, Ee, Ee, Ee, Ee,
      0, 0, 0, 0, 0, 0, 0, 1, 1.0f, 0, 0);

  // 14. x2 = LN(mha + dec_query)
  add_ln1024<<<Mr, 256, 0, stream>>>(da, dq, ln1_g, ln1_b, x2b);

  // 15. ff1 = relu(x2 @ W1^T + b1)
  gemm_wmma<<<dim3(16, 64, 1), 256, 0, stream>>>(
      x2b, W1, b1, ff1, Mr, Ff, Ee, Ee, Ee, Ff,
      0, 0, 0, 0, 0, 0, 0, 1, 1.0f, 1, 0);

  // 16. ff2 = ff1 @ W2^T + b2
  gemm_wmma<<<dim3(4, 64, 1), 256, 0, stream>>>(
      ff1, W2, b2, ff2, Mr, Ee, Ff, Ff, Ff, Ee,
      0, 0, 0, 0, 0, 0, 0, 1, 1.0f, 0, 0);

  // 17. out = LN(ff2 + x2)
  add_ln1024<<<Mr, 256, 0, stream>>>(ff2, x2b, ln2_g, ln2_b, out);
}
